// GraphMertMultiheadAttention_22024592294143
// MI455X (gfx1250) — matrix operand order repaired and checked by reading, not compiled, not run on hardware
//
#include <hip/hip_runtime.h>
#include <cstdint>
#include <cstddef>

// ---------------------------------------------------------------------------
// Types
// ---------------------------------------------------------------------------
using bf16 = __bf16;
typedef __attribute__((ext_vector_type(16))) __bf16 v16bf;
typedef __attribute__((ext_vector_type(8)))  float  v8f;
typedef __attribute__((ext_vector_type(4)))  unsigned int v4u;
typedef __attribute__((ext_vector_type(4)))  int    v4i;
typedef __attribute__((ext_vector_type(8)))  int    v8i;

// 32-byte WMMA operand fragment = two 16B chunks
union Frag {
  v16bf v;
  v4u   q[2];
};

__device__ inline v8f wmma_bf16(const v16bf a, const v16bf b, const v8f c) {
  // D = A(16x32 bf16) * B(32x16 bf16) + C(16x16 f32)
  return __builtin_amdgcn_wmma_f32_16x16x32_bf16(
      /*neg_a=*/false, a, /*neg_b=*/false, b,
      /*c_mod=*/(short)0, c, /*reuse_a=*/false, /*reuse_b=*/false);
}

// ---------------------------------------------------------------------------
// Problem constants
// ---------------------------------------------------------------------------
constexpr int T  = 1024;
constexpr int Bb = 4;
constexpr int H  = 1024;
constexpr int NH = 16;
constexpr int HD = 64;
constexpr int M  = T * Bb;   // 4096 rows for projections

// ---------------------------------------------------------------------------
// TDM availability (device pass on gfx1250 has it; host pass falls back)
// ---------------------------------------------------------------------------
#if __has_builtin(__builtin_amdgcn_tensor_load_to_lds)
#define HAVE_TDM 1
#else
#define HAVE_TDM 0
#endif

#if HAVE_TDM
// 2D tile DMA: tensor [tensorH rows x tensorW elems] of bf16 (data_size=2B),
// row stride rowStride elems; tile [tileH x tileW] -> LDS at byte offset ldsOff.
__device__ inline void tdm_load_tile(unsigned ldsOff, const void* gptr,
                                     unsigned tensorW, unsigned tensorH,
                                     unsigned long long rowStride,
                                     unsigned tileW, unsigned tileH) {
  unsigned long long ga = (unsigned long long)(uintptr_t)gptr;
  v4u g0;
  g0.x = 1u;                                     // count=1, user descriptor
  g0.y = ldsOff;                                 // lds_addr (bytes)
  g0.z = (unsigned)ga;                           // global_addr[31:0]
  g0.w = (unsigned)((ga >> 32) & 0x01FFFFFFu) | (2u << 30);  // [56:32] | type=2
  v8i g1;
  g1[0] = (int)(1u << 16);                       // wg_mask=0, data_size=1 (2B)
  g1[1] = (int)((tensorW & 0xFFFFu) << 16);      // tensor_dim0[15:0]
  g1[2] = (int)((tensorW >> 16) | ((tensorH & 0xFFFFu) << 16));
  g1[3] = (int)((tensorH >> 16) | (tileW << 16));// tensor_dim1 hi | tile_dim0
  g1[4] = (int)(tileH & 0xFFFFu);                // tile_dim1 (tile_dim2=0)
  g1[5] = (int)(unsigned)(rowStride & 0xFFFFFFFFull);
  g1[6] = (int)(unsigned)((rowStride >> 32) & 0xFFFFull);
  g1[7] = 0;
  v4i z4 = {0, 0, 0, 0};
#if defined(__clang_major__) && __clang_major__ >= 23
  v8i z8 = {0, 0, 0, 0, 0, 0, 0, 0};
  __builtin_amdgcn_tensor_load_to_lds(g0, g1, z4, z4, z8, 0);
#else
  __builtin_amdgcn_tensor_load_to_lds(g0, g1, z4, z4, 0);
#endif
}
#endif

// The TDM writes LDS behind the compiler's back (the descriptor carries the
// LDS address as plain bits, so alias analysis sees no store to smem and would
// fold the subsequent ds_loads to undef). Escape the tile pointer into opaque
// asm with a memory clobber (input-only, so the pointer itself keeps its
// LDS address-space provenance and the loads stay ds_load_b128, not flat).
__device__ inline void lds_fence(const void* p) {
  asm volatile("" : : "v"(p) : "memory");
}

// ---------------------------------------------------------------------------
// f32 -> bf16 conversion
// ---------------------------------------------------------------------------
__global__ __launch_bounds__(256) void cvt_f32_bf16(const float* __restrict__ in,
                                                    bf16* __restrict__ out, int n) {
  int i = blockIdx.x * 256 + threadIdx.x;
  if (i < n) out[i] = (bf16)in[i];
}

// ---------------------------------------------------------------------------
// GEMM: out = A[4096,1024] * W^T + bias   (A bf16 rowmajor, W[N,K] bf16 rowmajor)
// mode 0: bf16 out, head-split [bh, t, d]          (Q with scale, K)
// mode 1: bf16 out, head-split transposed [bh,d,t] (V)
// mode 2: f32 out, plain [m, n]                    (final projection)
// Block tile 128x64, 256 threads (8 waves as 4Mx2N), K-step 32, TDM double-buf.
// ---------------------------------------------------------------------------
__global__ __launch_bounds__(256) void gemm_bf16_nt(
    const bf16* __restrict__ A, const bf16* __restrict__ Bw,
    const float* __restrict__ bias, void* __restrict__ outp,
    float scale, int mode) {
  constexpr int K = 1024;
  constexpr int NK = K / 32;
  // A tile 128x32 bf16 = 8192B, B tile 64x32 bf16 = 4096B, double buffered
  __shared__ __align__(16) char smem[2 * 12288];

  const int bn = blockIdx.x;           // N/64 = 16
  const int bm = blockIdx.y;           // M/128 = 32
  const int tid = threadIdx.x;
  const int wid = tid >> 5;
  const int lane = tid & 31;
  const int r16 = lane & 15;
  const int hi = lane >> 4;
  const int wm = (wid >> 1) * 32;
  const int wn = (wid & 1) * 32;

  v8f c[2][2];
#pragma unroll
  for (int i = 0; i < 2; ++i)
#pragma unroll
    for (int j = 0; j < 2; ++j)
#pragma unroll
      for (int r = 0; r < 8; ++r) c[i][j][r] = 0.0f;

#if HAVE_TDM
  if (wid == 0) {   // one wave drives the TDM; EXEC is ignored by tensor ops
    tdm_load_tile(0u, (const char*)A + (size_t)bm * 128 * 2048,
                  1024u, 4096u, 1024ull, 32u, 128u);
    tdm_load_tile(8192u, (const char*)Bw + (size_t)bn * 64 * 2048,
                  1024u, 1024u, 1024ull, 32u, 64u);
  }
#endif

  for (int kt = 0; kt < NK; ++kt) {
#if HAVE_TDM
    const int cur = kt & 1;
    if (wid == 0) {
      if (kt + 1 < NK) {    // prefetch next K-tile into the other buffer
        const unsigned nb = (unsigned)((kt + 1) & 1) * 12288u;
        tdm_load_tile(nb, (const char*)A + (size_t)bm * 128 * 2048 + (size_t)(kt + 1) * 64,
                      1024u, 4096u, 1024ull, 32u, 128u);
        tdm_load_tile(nb + 8192u, (const char*)Bw + (size_t)bn * 64 * 2048 + (size_t)(kt + 1) * 64,
                      1024u, 1024u, 1024ull, 32u, 64u);
        __builtin_amdgcn_s_wait_tensorcnt(2);   // current tile's 2 DMAs done
      } else {
        __builtin_amdgcn_s_wait_tensorcnt(0);
      }
    }
    __syncthreads();
    const char* sA = smem + cur * 12288;   // stays addrspace(3)-inferable
    lds_fence(sA);                         // DMA wrote it; loads must happen
#else
    __syncthreads();
    for (int i = tid; i < 768; i += 256) {      // 768 x 16B chunks
      if (i < 512) {
        int row = i >> 2, p = i & 3;
        *(v4u*)(smem + row * 64 + p * 16) =
            *(const v4u*)((const char*)A + (size_t)(bm * 128 + row) * 2048 + kt * 64 + p * 16);
      } else {
        int j = i - 512, row = j >> 2, p = j & 3;
        *(v4u*)(smem + 8192 + row * 64 + p * 16) =
            *(const v4u*)((const char*)Bw + (size_t)(bn * 64 + row) * 2048 + kt * 64 + p * 16);
      }
    }
    __syncthreads();
    const char* sA = smem;
#endif
    const char* sB = sA + 8192;

    Frag a[2], b[2];
#pragma unroll
    for (int mt = 0; mt < 2; ++mt) {
      // A layout: lane<16 K{0..7,16..23}, lane>=16 K{8..15,24..31}
      const int bo = (wm + mt * 16 + r16) * 64 + hi * 16;
      a[mt].q[0] = *(const v4u*)(sA + bo);
      a[mt].q[1] = *(const v4u*)(sA + bo + 32);
    }
#pragma unroll
    for (int nt = 0; nt < 2; ++nt) {
      // B layout: lane<16 K{0..7,16..23}, lane>=16 K{8..15,24..31}
      const int bo = (wn + nt * 16 + r16) * 64 + hi * 16;
      b[nt].q[0] = *(const v4u*)(sB + bo);
      b[nt].q[1] = *(const v4u*)(sB + bo + 32);
    }
#pragma unroll
    for (int mt = 0; mt < 2; ++mt)
#pragma unroll
      for (int nt = 0; nt < 2; ++nt)
        c[mt][nt] = wmma_bf16(a[mt].v, b[nt].v, c[mt][nt]);

#if HAVE_TDM
    __syncthreads();   // release current buffer before next prefetch overwrites
#endif
  }

  // Epilogue: C layout lane->(M = r + 8*hi, N = r16)
#pragma unroll
  for (int mt = 0; mt < 2; ++mt) {
#pragma unroll
    for (int nt = 0; nt < 2; ++nt) {
      const int gn = bn * 64 + wn + nt * 16 + r16;
      const float bv = bias[gn];
#pragma unroll
      for (int r = 0; r < 8; ++r) {
        const int gm = bm * 128 + wm + mt * 16 + r + 8 * hi;
        const float val = (c[mt][nt][r] + bv) * scale;
        if (mode == 2) {
          ((float*)outp)[(size_t)gm * 1024 + gn] = val;
        } else {
          const int t = gm >> 2, bb = gm & 3;          // m = t*B + b
          const int nh = gn >> 6, d = gn & 63;
          const int bh = bb * 16 + nh;
          if (mode == 0)
            ((bf16*)outp)[((size_t)bh * 1024 + t) * 64 + d] = (bf16)val;
          else
            ((bf16*)outp)[((size_t)bh * 64 + d) * 1024 + t] = (bf16)val;
        }
      }
    }
  }
}

// ---------------------------------------------------------------------------
// Flash attention: per head bh, per 64-row q block (4 waves x 16 rows).
// scores = (Q K^T + bias) * decay; online softmax; acc += P V.
// Q,K: [bh][t][64] bf16.  Vt: [bh][64][t] bf16.  out: [t*B+b][H] bf16.
// ---------------------------------------------------------------------------
__global__ __launch_bounds__(128) void flash_attn(
    const bf16* __restrict__ Q, const bf16* __restrict__ Km,
    const bf16* __restrict__ Vt, const float* __restrict__ bias,
    const float* __restrict__ decay, bf16* __restrict__ outTB) {
  const int bh = blockIdx.y;                      // 64 heads*batch
  const int wid = threadIdx.x >> 5;               // 4 waves
  const int lane = threadIdx.x & 31;
  const int r16 = lane & 15;
  const int hi = lane >> 4;
  const int t0 = blockIdx.x * 64 + wid * 16;

  const bf16* Qh = Q + (size_t)bh * 1024 * 64;
  const bf16* Kh = Km + (size_t)bh * 1024 * 64;
  const bf16* Vh = Vt + (size_t)bh * 64 * 1024;
  const float* biasH = bias + (size_t)bh * 1024 * 1024;
  const float* decH = decay + (size_t)bh * 1024 * 1024;

  __shared__ __align__(16) bf16 Pbuf[4][16 * 32];  // per-wave P tile (bf16)
  bf16* myP = Pbuf[wid];

  // Q fragments (A layout), rows t0..t0+15, K=64 split in two 32-chunks
  Frag aQ[2];
#pragma unroll
  for (int ks = 0; ks < 2; ++ks) {
    const int bo = (t0 + r16) * 128 + ks * 64 + hi * 16;
    aQ[ks].q[0] = *(const v4u*)((const char*)Qh + bo);
    aQ[ks].q[1] = *(const v4u*)((const char*)Qh + bo + 32);
  }

  v8f acc[4];
  float mstat[8], lstat[8];
#pragma unroll
  for (int dt = 0; dt < 4; ++dt)
#pragma unroll
    for (int r = 0; r < 8; ++r) acc[dt][r] = 0.0f;
#pragma unroll
  for (int r = 0; r < 8; ++r) { mstat[r] = -3.0e38f; lstat[r] = 0.0f; }

  for (int s0 = 0; s0 < 1024; s0 += 32) {
    // ---- two 16x16 score tiles (s0, s0+16), K=64 over 2 WMMAs each ----
    v8f sv[2];
#pragma unroll
    for (int st = 0; st < 2; ++st) {
      v8f ct;
#pragma unroll
      for (int r = 0; r < 8; ++r) ct[r] = 0.0f;
#pragma unroll
      for (int ks = 0; ks < 2; ++ks) {
        Frag bK;  // B layout: lane<16 K{0..7,16..23} (d), lane>=16 K{8..15,24..31}
        const int row = s0 + st * 16 + r16;
        const int bo = row * 128 + ks * 64 + hi * 16;
        bK.q[0] = *(const v4u*)((const char*)Kh + bo);
        bK.q[1] = *(const v4u*)((const char*)Kh + bo + 32);
        ct = wmma_bf16(aQ[ks].v, bK.v, ct);
      }
      // fused bias add + decay mul, exact C-fragment coordinates.
      // bias/decay are streamed exactly once (512MB total) -> non-temporal
      // so they don't evict the K/V working set from L2.
#pragma unroll
      for (int r = 0; r < 8; ++r) {
        const size_t idx = (size_t)(t0 + r + 8 * hi) * 1024 + s0 + st * 16 + r16;
        const float bvl = __builtin_nontemporal_load(&biasH[idx]);
        const float dvl = __builtin_nontemporal_load(&decH[idx]);
        ct[r] = (ct[r] + bvl) * dvl;
      }
      sv[st] = ct;
    }

    // ---- online softmax, rows live as vgpr r within 16-lane halves ----
    float pscale[8];
#pragma unroll
    for (int r = 0; r < 8; ++r) {
      float mx = fmaxf(sv[0][r], sv[1][r]);
      mx = fmaxf(mx, __shfl_xor(mx, 1));
      mx = fmaxf(mx, __shfl_xor(mx, 2));
      mx = fmaxf(mx, __shfl_xor(mx, 4));
      mx = fmaxf(mx, __shfl_xor(mx, 8));
      const float mnew = fmaxf(mstat[r], mx);
      const float rs = __expf(mstat[r] - mnew);
      mstat[r] = mnew;
      const float p0 = __expf(sv[0][r] - mnew);
      const float p1 = __expf(sv[1][r] - mnew);
      sv[0][r] = p0;
      sv[1][r] = p1;
      float ps = p0 + p1;
      ps += __shfl_xor(ps, 1);
      ps += __shfl_xor(ps, 2);
      ps += __shfl_xor(ps, 4);
      ps += __shfl_xor(ps, 8);
      lstat[r] = lstat[r] * rs + ps;
      pscale[r] = rs;
    }
#pragma unroll
    for (int dt = 0; dt < 4; ++dt)
#pragma unroll
      for (int r = 0; r < 8; ++r) acc[dt][r] *= pscale[r];

    // ---- P (C layout) -> LDS -> A fragment (layout change) ----
#pragma unroll
    for (int r = 0; r < 8; ++r) {
      myP[(r + 8 * hi) * 32 + r16] = (bf16)sv[0][r];
      myP[(r + 8 * hi) * 32 + 16 + r16] = (bf16)sv[1][r];
    }
    Frag aP;
    {
      const int bo = r16 * 64 + hi * 16;
      aP.q[0] = *(const v4u*)((const char*)myP + bo);
      aP.q[1] = *(const v4u*)((const char*)myP + bo + 32);
    }

    // ---- acc += P[16x32] * V[32x64]  (Vt rows are d, s-contiguous) ----
#pragma unroll
    for (int dt = 0; dt < 4; ++dt) {
      Frag bV;
      const int row = dt * 16 + r16;
      const int bo = row * 2048 + s0 * 2 + hi * 16;
      bV.q[0] = *(const v4u*)((const char*)Vh + bo);
      bV.q[1] = *(const v4u*)((const char*)Vh + bo + 32);
      acc[dt] = wmma_bf16(aP.v, bV.v, acc[dt]);
    }
  }

  // ---- normalize and store back in [t*B+b][H] bf16 for output GEMM ----
  const int b = bh >> 4, nh = bh & 15;
#pragma unroll
  for (int dt = 0; dt < 4; ++dt) {
#pragma unroll
    for (int r = 0; r < 8; ++r) {
      const int t = t0 + r + 8 * hi;
      const int col = nh * 64 + dt * 16 + r16;
      outTB[(size_t)(t * 4 + b) * 1024 + col] = (bf16)(acc[dt][r] / lstat[r]);
    }
  }
}

// ---------------------------------------------------------------------------
// Workspace layout (bytes) — total 48 MB
// ---------------------------------------------------------------------------
constexpr size_t OFF_XBF = 0;                         // query bf16 [4096,1024]  8MB
constexpr size_t OFF_WQ = 8388608;                    // 2MB each
constexpr size_t OFF_WK = OFF_WQ + 2097152;
constexpr size_t OFF_WV = OFF_WK + 2097152;
constexpr size_t OFF_WO = OFF_WV + 2097152;
constexpr size_t OFF_Q = OFF_WO + 2097152;            // [64][1024][64] bf16 8MB
constexpr size_t OFF_K = OFF_Q + 8388608;
constexpr size_t OFF_VT = OFF_K + 8388608;            // [64][64][1024] bf16 8MB
constexpr size_t OFF_ATT = OFF_VT + 8388608;          // [4096][1024] bf16 8MB

extern "C" void kernel_launch(void* const* d_in, const int* in_sizes, int n_in,
                              void* d_out, int out_size, void* d_ws, size_t ws_size,
                              hipStream_t stream) {
  (void)in_sizes; (void)n_in; (void)out_size; (void)ws_size;
  const float* query = (const float*)d_in[0];
  const float* decay = (const float*)d_in[1];
  const float* abias = (const float*)d_in[2];
  const float* Wq = (const float*)d_in[3];
  const float* bq = (const float*)d_in[4];
  const float* Wk = (const float*)d_in[5];
  const float* bk = (const float*)d_in[6];
  const float* Wv = (const float*)d_in[7];
  const float* bv = (const float*)d_in[8];
  const float* Wo = (const float*)d_in[9];
  const float* bo = (const float*)d_in[10];

  char* ws = (char*)d_ws;
  bf16* Xbf = (bf16*)(ws + OFF_XBF);
  bf16* Wqb = (bf16*)(ws + OFF_WQ);
  bf16* Wkb = (bf16*)(ws + OFF_WK);
  bf16* Wvb = (bf16*)(ws + OFF_WV);
  bf16* Wob = (bf16*)(ws + OFF_WO);
  bf16* Qbf = (bf16*)(ws + OFF_Q);
  bf16* Kbf = (bf16*)(ws + OFF_K);
  bf16* Vtb = (bf16*)(ws + OFF_VT);
  bf16* Att = (bf16*)(ws + OFF_ATT);

  // 1) f32 -> bf16 packs
  cvt_f32_bf16<<<dim3(16384), dim3(256), 0, stream>>>(query, Xbf, M * H);
  cvt_f32_bf16<<<dim3(4096), dim3(256), 0, stream>>>(Wq, Wqb, H * H);
  cvt_f32_bf16<<<dim3(4096), dim3(256), 0, stream>>>(Wk, Wkb, H * H);
  cvt_f32_bf16<<<dim3(4096), dim3(256), 0, stream>>>(Wv, Wvb, H * H);
  cvt_f32_bf16<<<dim3(4096), dim3(256), 0, stream>>>(Wo, Wob, H * H);

  // 2) QKV projections (fused bias, q-scaling, head split, V transpose)
  const float scaling = 0.125f;  // 1/sqrt(HD)
  gemm_bf16_nt<<<dim3(16, 32), dim3(256), 0, stream>>>(Xbf, Wqb, bq, Qbf, scaling, 0);
  gemm_bf16_nt<<<dim3(16, 32), dim3(256), 0, stream>>>(Xbf, Wkb, bk, Kbf, 1.0f, 0);
  gemm_bf16_nt<<<dim3(16, 32), dim3(256), 0, stream>>>(Xbf, Wvb, bv, Vtb, 1.0f, 1);

  // 3) fused attention (bias add + decay mul + online softmax + PV)
  flash_attn<<<dim3(16, 64), dim3(128), 0, stream>>>(Qbf, Kbf, Vtb, abias, decay, Att);

  // 4) output projection -> f32 d_out
  gemm_bf16_nt<<<dim3(16, 32), dim3(256), 0, stream>>>(Att, Wob, bo, d_out, 1.0f, 2);
}
